// PointTransformerBlock_46840913330605
// MI455X (gfx1250) — matrix operand order repaired and checked by reading, not compiled, not run on hardware
//
#include <hip/hip_runtime.h>

// ---------------------------------------------------------------------------
// Point Transformer block for MI455X (gfx1250, wave32, WMMA).
// Shapes: B=4, N=8192, F=128, C=64, K=16.
//
// All dense GEMMs use v_wmma_f32_16x16x32_f16 (f16 in, f32 accumulate).
// Fragment layouts per CDNA5 ISA §7.12.2:
//   A (16x32 f16): lane l holds row M=l%16; half h -> K = (l>=16?8:0) + (h<8?h:h+8)
//   B (32x16 f16): lane l holds col N=l%16; half h -> K = (l>=16?8:0) + (h<8?h:h+8)
//   C/D (16x16 f32): lane l, float r -> (M = r + 8*(l>>4), N = l%16)
//
// Weights (~80KB total) are converted to f16 and swizzled into per-lane
// B-fragment order ONCE by a prep kernel; GEMMs then read every B fragment as
// 2x global_load_b128 straight from the L2/WGP$-resident swizzled copy.
// GEMM kernels use no LDS and no barriers.
// ---------------------------------------------------------------------------

typedef __attribute__((ext_vector_type(16))) _Float16 v16h;
typedef __attribute__((ext_vector_type(2)))  __fp16   fp16x2;   // cvt_pkrtz result
typedef __attribute__((ext_vector_type(8)))  float    v8f;

#define BATCH 4
#define NPTS  8192
#define BN    32768      // BATCH * NPTS
#define FDIM  128
#define CDIM  64
#define KNN   16

// Swizzled-weight region offsets (in halfs) inside the workspace.
#define WZ_ENC 0         // 128x64 -> 8192
#define WZ_Q   8192      // 64x64  -> 4096
#define WZ_K   12288
#define WZ_V   16384
#define WZ_P2  20480
#define WZ_A1  24576
#define WZ_A2  28672
#define WZ_DEC 32768     // 64x128 -> 8192
#define WZ_TOTAL 40960

__device__ __forceinline__ int a_k_of(int lane, int h) {
  return ((lane >= 16) ? 8 : 0) + ((h < 8) ? h : h + 8);
}

// Flat index of element (row mm, channel c) inside the A-fragment-ready
// layout [frag f][lane 0..31][half 0..15] (inverse of a_k_of).
__device__ __forceinline__ int afrag_addr(int mm, int c) {
  int f2  = c >> 5;
  int kk  = c & 31;
  int hi2 = (kk >> 3) & 1;                       // k 8..15 / 24..31 -> upper lanes
  int h2  = (kk & 7) + ((kk >= 16) ? 8 : 0);
  return (f2 * 32 + mm + 16 * hi2) * 16 + h2;
}

// Convert W[K,N] (f32 row-major) to f16 permuted into B-fragment order:
// element e = ((ks*NNT + nt)*32 + lane)*16 + h.
template <int N, int K>
__device__ __forceinline__ void swz_weights(_Float16* __restrict__ dst,
                                            const float* __restrict__ W, int tid) {
  constexpr int NNT = N >> 4;
  for (int e = tid; e < K * N; e += 256) {
    int h  = e & 15;
    int l2 = (e >> 4) & 31;
    int t2 = e >> 9;
    int nt = t2 % NNT;
    int ks = t2 / NNT;
    int k  = ks * 32 + ((l2 >= 16) ? 8 : 0) + ((h < 8) ? h : h + 8);   // b_k_of
    int n  = nt * 16 + (l2 & 15);
    dst[e] = (_Float16)W[k * N + n];
  }
}

// One-time weight prep: one matrix per block.
__global__ __launch_bounds__(256)
void prep_weights_kernel(const float* __restrict__ W_enc, const float* __restrict__ Wq,
                         const float* __restrict__ Wk,    const float* __restrict__ Wv,
                         const float* __restrict__ Wp2,   const float* __restrict__ Wa1,
                         const float* __restrict__ Wa2,   const float* __restrict__ W_dec,
                         _Float16* __restrict__ wz) {
  int tid = threadIdx.x;
  switch (blockIdx.x) {
    case 0: swz_weights<CDIM, FDIM>(wz + WZ_ENC, W_enc, tid); break;
    case 1: swz_weights<CDIM, CDIM>(wz + WZ_Q,  Wq,  tid); break;
    case 2: swz_weights<CDIM, CDIM>(wz + WZ_K,  Wk,  tid); break;
    case 3: swz_weights<CDIM, CDIM>(wz + WZ_V,  Wv,  tid); break;
    case 4: swz_weights<CDIM, CDIM>(wz + WZ_P2, Wp2, tid); break;
    case 5: swz_weights<CDIM, CDIM>(wz + WZ_A1, Wa1, tid); break;
    case 6: swz_weights<CDIM, CDIM>(wz + WZ_A2, Wa2, tid); break;
    case 7: swz_weights<FDIM, CDIM>(wz + WZ_DEC, W_dec, tid); break;
  }
}

// Load one A fragment (16x32) from an f32 row: two contiguous 8-float runs,
// vectorized b128 loads + packed f16 converts (v_cvt_pk_rtz_f16_f32).
__device__ __forceinline__ v16h load_a_frag_f32(const float* __restrict__ row,
                                                int lane, int ksbase) {
  int lo = (lane >= 16) ? 8 : 0;
  const float4* p = (const float4*)(row + ksbase + lo);
  float4 x0 = p[0], x1 = p[1];     // k = base+lo .. +7
  float4 x2 = p[4], x3 = p[5];     // k = base+lo+16 .. +23
  union { v16h v; fp16x2 p2[8]; } u;
  u.p2[0] = __builtin_amdgcn_cvt_pkrtz(x0.x, x0.y);
  u.p2[1] = __builtin_amdgcn_cvt_pkrtz(x0.z, x0.w);
  u.p2[2] = __builtin_amdgcn_cvt_pkrtz(x1.x, x1.y);
  u.p2[3] = __builtin_amdgcn_cvt_pkrtz(x1.z, x1.w);
  u.p2[4] = __builtin_amdgcn_cvt_pkrtz(x2.x, x2.y);
  u.p2[5] = __builtin_amdgcn_cvt_pkrtz(x2.z, x2.w);
  u.p2[6] = __builtin_amdgcn_cvt_pkrtz(x3.x, x3.y);
  u.p2[7] = __builtin_amdgcn_cvt_pkrtz(x3.z, x3.w);
  return u.v;
}

// ---------------------------------------------------------------------------
// WMMA GEMM: C[M,N] = A[M,K] * Wz + bias[N] (+ resid[M,N]).
// No LDS, no barriers; B fragments straight from the swizzled global copy.
// 256 threads = 8 waves, 128 rows per block.
// ---------------------------------------------------------------------------
template <int N, int K, bool RESID>
__global__ __launch_bounds__(256)
void gemm_wmma_kernel(const float* __restrict__ A, const _Float16* __restrict__ Wz,
                      const float* __restrict__ bias, const float* __restrict__ resid,
                      float* __restrict__ C) {
  constexpr int NK  = K >> 5;
  constexpr int NNT = N >> 4;
  int tid = threadIdx.x;
  int lane = tid & 31, wave = tid >> 5;
  int row0 = blockIdx.x * 128 + wave * 16;
  int m  = lane & 15;
  int hi = lane >> 4;
  const float* arow = A + (size_t)(row0 + m) * K;
  __builtin_prefetch(arow, 0, 1);                 // global_prefetch_b8

  v16h afr[NK];
  #pragma unroll
  for (int ks = 0; ks < NK; ++ks) afr[ks] = load_a_frag_f32(arow, lane, ks * 32);

  const v16h* wf = (const v16h*)Wz + lane;        // frag f at wf[f*32]
  #pragma unroll
  for (int nt = 0; nt < NNT; ++nt) {
    int col = nt * 16 + m;
    float bv = bias[col];
    v8f acc;
    #pragma unroll
    for (int r = 0; r < 8; ++r) acc[r] = bv;
    #pragma unroll
    for (int ks = 0; ks < NK; ++ks) {
      v16h bf = wf[(ks * NNT + nt) * 32];
      acc = __builtin_amdgcn_wmma_f32_16x16x32_f16(false, afr[ks], false, bf,
                                                   (short)0, acc, false, false);
    }
    size_t base = (size_t)(row0 + 8 * hi) * N + col;
    float* cp = C + base;
    if (RESID) {
      const float* rp = resid + base;
      #pragma unroll
      for (int r = 0; r < 8; ++r) cp[r * N] = acc[r] + rp[r * N];
    } else {
      #pragma unroll
      for (int r = 0; r < 8; ++r) cp[r * N] = acc[r];
    }
  }
}

// ---------------------------------------------------------------------------
// Fused q/k/v projection: x loaded once; Wq/Wk/Wv contiguous in wz.
// ---------------------------------------------------------------------------
__global__ __launch_bounds__(256)
void qkv_wmma_kernel(const float* __restrict__ X, const _Float16* __restrict__ WzQ,
                     const float* __restrict__ bq, const float* __restrict__ bk,
                     const float* __restrict__ bv,
                     float* __restrict__ qo, float* __restrict__ ko,
                     float* __restrict__ vo) {
  int tid = threadIdx.x;
  int lane = tid & 31, wave = tid >> 5;
  int row0 = blockIdx.x * 128 + wave * 16;
  int m  = lane & 15;
  int hi = lane >> 4;
  const float* arow = X + (size_t)(row0 + m) * CDIM;
  __builtin_prefetch(arow, 0, 1);

  v16h afr[2];
  #pragma unroll
  for (int ks = 0; ks < 2; ++ks) afr[ks] = load_a_frag_f32(arow, lane, ks * 32);

  const v16h* wf = (const v16h*)WzQ + lane;       // q:0..7, k:8..15, v:16..23 frags
  #pragma unroll
  for (int w = 0; w < 3; ++w) {
    float* O = (w == 0) ? qo : ((w == 1) ? ko : vo);
    const float* bias = (w == 0) ? bq : ((w == 1) ? bk : bv);
    #pragma unroll
    for (int nt = 0; nt < 4; ++nt) {
      int col = nt * 16 + m;
      float bvv = bias[col];
      v8f acc;
      #pragma unroll
      for (int r = 0; r < 8; ++r) acc[r] = bvv;
      #pragma unroll
      for (int ks = 0; ks < 2; ++ks) {
        v16h bf = wf[(w * 8 + ks * 4 + nt) * 32];
        acc = __builtin_amdgcn_wmma_f32_16x16x32_f16(false, afr[ks], false, bf,
                                                     (short)0, acc, false, false);
      }
      float* cp = O + (size_t)(row0 + 8 * hi) * CDIM + col;
      #pragma unroll
      for (int r = 0; r < 8; ++r) cp[r * CDIM] = acc[r];
    }
  }
}

// ---------------------------------------------------------------------------
// Brute-force kNN per point cloud; top-16 via unrolled register swap network.
// ---------------------------------------------------------------------------
__global__ __launch_bounds__(256)
void knn_kernel(const float* __restrict__ pos, int* __restrict__ idxOut) {
  __shared__ float sp[256 * 3];
  int b = blockIdx.y;
  int i = blockIdx.x * 256 + threadIdx.x;
  const float* P = pos + (size_t)b * NPTS * 3;
  float px = P[i * 3 + 0], py = P[i * 3 + 1], pz = P[i * 3 + 2];

  float bd[KNN]; int bi[KNN];
  #pragma unroll
  for (int t = 0; t < KNN; ++t) { bd[t] = 3.4e38f; bi[t] = 0; }

  for (int jt = 0; jt < NPTS; jt += 256) {
    __syncthreads();
    int j = jt + threadIdx.x;
    sp[threadIdx.x * 3 + 0] = P[j * 3 + 0];
    sp[threadIdx.x * 3 + 1] = P[j * 3 + 1];
    sp[threadIdx.x * 3 + 2] = P[j * 3 + 2];
    __syncthreads();
    for (int jj = 0; jj < 256; ++jj) {
      float dx = px - sp[jj * 3 + 0];
      float dy = py - sp[jj * 3 + 1];
      float dz = pz - sp[jj * 3 + 2];
      float d = dx * dx + dy * dy + dz * dz;
      if (d < bd[KNN - 1]) {
        float cd = d; int ci = jt + jj;
        #pragma unroll
        for (int t = 0; t < KNN; ++t) {
          bool sm = cd < bd[t];
          float td = bd[t]; int ti = bi[t];
          bd[t] = sm ? cd : td;  bi[t] = sm ? ci : ti;
          cd    = sm ? td : cd;  ci    = sm ? ti : ci;
        }
      }
    }
  }
  #pragma unroll
  for (int t = 0; t < KNN; ++t)
    idxOut[((size_t)b * NPTS + i) * KNN + t] = bi[t];
}

// h1 helper: 8 consecutive channels of relu(rel@Wp1+bp1), vectorized float4
// reads of Wp1 rows + packed converts. cb is the channel base (multiple of 8).
__device__ __forceinline__ void h1_run8(float rx, float ry, float rz,
                                        const float* __restrict__ Wp1,
                                        const float* __restrict__ bp1,
                                        int cb, fp16x2* out) {
  const float4* w0 = (const float4*)(Wp1 + cb);
  const float4* w1 = (const float4*)(Wp1 + 64 + cb);
  const float4* w2 = (const float4*)(Wp1 + 128 + cb);
  const float4* bb = (const float4*)(bp1 + cb);
  #pragma unroll
  for (int q = 0; q < 2; ++q) {
    float4 a = w0[q], b = w1[q], c = w2[q], d = bb[q];
    float e0 = fmaxf(rx * a.x + ry * b.x + rz * c.x + d.x, 0.0f);
    float e1 = fmaxf(rx * a.y + ry * b.y + rz * c.y + d.y, 0.0f);
    float e2 = fmaxf(rx * a.z + ry * b.z + rz * c.z + d.z, 0.0f);
    float e3 = fmaxf(rx * a.w + ry * b.w + rz * c.w + d.w, 0.0f);
    out[2 * q + 0] = __builtin_amdgcn_cvt_pkrtz(e0, e1);
    out[2 * q + 1] = __builtin_amdgcn_cvt_pkrtz(e2, e3);
  }
}

// ---------------------------------------------------------------------------
// Fused per-point attention: one wave per point (16 neighbors == WMMA M=16).
// LDS only for per-wave restriping tile / idx / rel; barrier-free.
// ---------------------------------------------------------------------------
__global__ __launch_bounds__(256)
void attn_kernel(const float* __restrict__ pos,
                 const float* __restrict__ qg, const float* __restrict__ kg,
                 const float* __restrict__ vg, const int* __restrict__ nidx,
                 const _Float16* __restrict__ WzP2,     // p2/a1/a2 contiguous
                 const float* __restrict__ Wp1, const float* __restrict__ bp1,
                 const float* __restrict__ bp2, const float* __restrict__ ba1,
                 const float* __restrict__ ba2, float* __restrict__ y) {
  __shared__ __align__(32) _Float16 sT[8][1024];   // per-wave A-frag-layout tile
  __shared__ int   sIdx[8][16];
  __shared__ float sRel[8][16][3];

  int tid = threadIdx.x;
  int lane = tid & 31, wave = tid >> 5;
  int pt = blockIdx.x * 8 + wave;            // global point id in [0, B*N)
  int b  = pt / NPTS, ii = pt % NPTS;
  const float* Pb = pos + (size_t)b * NPTS * 3;

  if (lane < 16) {
    int j = nidx[(size_t)pt * KNN + lane];
    sIdx[wave][lane] = j;
    sRel[wave][lane][0] = Pb[ii * 3 + 0] - Pb[j * 3 + 0];
    sRel[wave][lane][1] = Pb[ii * 3 + 1] - Pb[j * 3 + 1];
    sRel[wave][lane][2] = Pb[ii * 3 + 2] - Pb[j * 3 + 2];
  }
  // Wave-internal DS ops are hardware-ordered (DScnt, in-order per wave).
  int m  = lane & 15;
  int hi = lane >> 4;
  int lo = (lane >= 16) ? 8 : 0;
  float rx = sRel[wave][m][0], ry = sRel[wave][m][1], rz = sRel[wave][m][2];

  // h1 = relu(rel @ Wp1 + bp1), built directly in A-fragment layout (K=3)
  v16h h1f[2];
  #pragma unroll
  for (int f = 0; f < 2; ++f) {
    union { v16h v; fp16x2 p2[8]; } u;
    h1_run8(rx, ry, rz, Wp1, bp1, f * 32 + lo,      u.p2);      // h 0..7
    h1_run8(rx, ry, rz, Wp1, bp1, f * 32 + lo + 16, u.p2 + 4);  // h 8..15
    h1f[f] = u.v;
  }

  const v16h* wf = (const v16h*)WzP2 + lane;  // p2: frags 0..7, a1: 8..15, a2: 16..23

  // delta = h1 @ Wp2 + bp2  (D layout kept in regs for later reuse)
  v8f dlt[4];
  #pragma unroll
  for (int nt = 0; nt < 4; ++nt) {
    int c = nt * 16 + m;
    float bb = bp2[c];
    v8f acc;
    #pragma unroll
    for (int r = 0; r < 8; ++r) acc[r] = bb;
    #pragma unroll
    for (int ks = 0; ks < 2; ++ks) {
      v16h bf = wf[(ks * 4 + nt) * 32];
      acc = __builtin_amdgcn_wmma_f32_16x16x32_f16(false, h1f[ks], false, bf,
                                                   (short)0, acc, false, false);
    }
    dlt[nt] = acc;
  }

  // t = q_i - k_j + delta -> store straight into A-fragment-layout LDS tile
  #pragma unroll
  for (int nt = 0; nt < 4; ++nt) {
    int c = nt * 16 + m;
    float qv = qg[(size_t)pt * CDIM + c];
    #pragma unroll
    for (int r = 0; r < 8; ++r) {
      int mm = r + 8 * hi;
      int j = sIdx[wave][mm];
      float kv = kg[((size_t)b * NPTS + j) * CDIM + c];
      sT[wave][afrag_addr(mm, c)] = (_Float16)(qv - kv + dlt[nt][r]);
    }
  }
  v16h tf[2];
  #pragma unroll
  for (int f = 0; f < 2; ++f)
    tf[f] = *(const v16h*)(sT[wave] + (f * 32 + lane) * 16);

  // h2 = relu(t @ Wa1 + ba1) -> back into A-fragment-layout tile
  #pragma unroll
  for (int nt = 0; nt < 4; ++nt) {
    int c = nt * 16 + m;
    float bb = ba1[c];
    v8f acc;
    #pragma unroll
    for (int r = 0; r < 8; ++r) acc[r] = bb;
    #pragma unroll
    for (int ks = 0; ks < 2; ++ks) {
      v16h bf = wf[(8 + ks * 4 + nt) * 32];
      acc = __builtin_amdgcn_wmma_f32_16x16x32_f16(false, tf[ks], false, bf,
                                                   (short)0, acc, false, false);
    }
    #pragma unroll
    for (int r = 0; r < 8; ++r)
      sT[wave][afrag_addr(r + 8 * hi, c)] = (_Float16)fmaxf(acc[r], 0.0f);
  }
  v16h gf[2];
  #pragma unroll
  for (int f = 0; f < 2; ++f)
    gf[f] = *(const v16h*)(sT[wave] + (f * 32 + lane) * 16);

  // a = h2 @ Wa2 + ba2
  v8f aacc[4];
  #pragma unroll
  for (int nt = 0; nt < 4; ++nt) {
    int c = nt * 16 + m;
    float bb = ba2[c];
    v8f acc;
    #pragma unroll
    for (int r = 0; r < 8; ++r) acc[r] = bb;
    #pragma unroll
    for (int ks = 0; ks < 2; ++ks) {
      v16h bf = wf[(16 + ks * 4 + nt) * 32];
      acc = __builtin_amdgcn_wmma_f32_16x16x32_f16(false, gf[ks], false, bf,
                                                   (short)0, acc, false, false);
    }
    aacc[nt] = acc;
  }

  // softmax over 16 neighbors per channel; lanes l and l^16 hold the two
  // 8-row halves of each column -> combine via shfl_xor(16).
  #pragma unroll
  for (int nt = 0; nt < 4; ++nt) {
    int c = nt * 16 + m;
    float mx = aacc[nt][0];
    #pragma unroll
    for (int r = 1; r < 8; ++r) mx = fmaxf(mx, aacc[nt][r]);
    mx = fmaxf(mx, __shfl_xor(mx, 16, 32));
    float s = 0.0f, wsum = 0.0f;
    #pragma unroll
    for (int r = 0; r < 8; ++r) {
      float e = __expf(aacc[nt][r] - mx);
      s += e;
      int mm = r + 8 * hi;
      int j = sIdx[wave][mm];
      float vval = vg[((size_t)b * NPTS + j) * CDIM + c];
      wsum += e * (vval + dlt[nt][r]);
    }
    s    += __shfl_xor(s, 16, 32);
    wsum += __shfl_xor(wsum, 16, 32);
    if (hi == 0) y[(size_t)pt * CDIM + c] = wsum / s;
  }
}

// ---------------------------------------------------------------------------
extern "C" void kernel_launch(void* const* d_in, const int* in_sizes, int n_in,
                              void* d_out, int out_size, void* d_ws, size_t ws_size,
                              hipStream_t stream) {
  const float* features  = (const float*)d_in[0];
  const float* positions = (const float*)d_in[1];
  const void*  batch     = d_in[2];
  // d_in[3] = num_neighbors (compile-time 16)
  const float* W_enc = (const float*)d_in[4];
  const float* b_enc = (const float*)d_in[5];
  const float* Wq = (const float*)d_in[6],  *bq = (const float*)d_in[7];
  const float* Wk = (const float*)d_in[8],  *bk = (const float*)d_in[9];
  const float* Wv = (const float*)d_in[10], *bv = (const float*)d_in[11];
  const float* Wp1 = (const float*)d_in[12], *bp1 = (const float*)d_in[13];
  const float* Wp2 = (const float*)d_in[14], *bp2 = (const float*)d_in[15];
  const float* Wa1 = (const float*)d_in[16], *ba1 = (const float*)d_in[17];
  const float* Wa2 = (const float*)d_in[18], *ba2 = (const float*)d_in[19];
  const float* W_dec = (const float*)d_in[20], *b_dec = (const float*)d_in[21];

  float* out = (float*)d_out;
  float* ws  = (float*)d_ws;
  const size_t XC = (size_t)BN * CDIM;      // 2,097,152 floats per buffer
  float* xb = ws;
  float* qb = ws + 1 * XC;
  float* kb = ws + 2 * XC;
  float* vb = ws + 3 * XC;
  float* yb = ws + 4 * XC;
  int*   idx = (int*)(ws + 5 * XC);                 // BN*16 ints
  _Float16* wz = (_Float16*)(ws + 5 * XC + (size_t)BN * KNN);  // 40960 halfs

  // 0) one-time weight convert+swizzle (L2-resident thereafter)
  prep_weights_kernel<<<8, 256, 0, stream>>>(W_enc, Wq, Wk, Wv, Wp2, Wa1, Wa2,
                                             W_dec, wz);
  // 1) encode: x = features @ W_enc + b_enc           [BN,128]x[128,64]
  gemm_wmma_kernel<CDIM, FDIM, false><<<BN / 128, 256, 0, stream>>>(
      features, wz + WZ_ENC, b_enc, nullptr, xb);
  // 2) fused q/k/v projections                        [BN,64]x[64,64] x3
  qkv_wmma_kernel<<<BN / 128, 256, 0, stream>>>(xb, wz + WZ_Q, bq, bk, bv,
                                                qb, kb, vb);
  // 3) kNN per point cloud
  knn_kernel<<<dim3(NPTS / 256, BATCH), 256, 0, stream>>>(positions, idx);
  // 4) fused per-point vector attention (one wave per point)
  attn_kernel<<<BN / 8, 256, 0, stream>>>(positions, qb, kb, vb, idx,
                                          wz + WZ_P2, Wp1, bp1, bp2, ba1, ba2, yb);
  // 5) decode + residual: out = y @ W_dec + b_dec + features
  gemm_wmma_kernel<FDIM, CDIM, true><<<BN / 128, 256, 0, stream>>>(
      yb, wz + WZ_DEC, b_dec, features, out);
  // 6) tuple tail: positions and batch pass-through
  (void)hipMemcpyAsync(out + (size_t)BN * FDIM, positions,
                       (size_t)BN * 3 * sizeof(float), hipMemcpyDeviceToDevice, stream);
  (void)hipMemcpyAsync(out + (size_t)BN * FDIM + (size_t)BN * 3, batch,
                       (size_t)BN * sizeof(int), hipMemcpyDeviceToDevice, stream);
  (void)in_sizes; (void)n_in; (void)out_size; (void)ws_size;
}
